// SpikeToGraph1D_11003706212836
// MI455X (gfx1250) — matrix operand order repaired and checked by reading, not compiled, not run on hardware
//
#include <hip/hip_runtime.h>
#include <math.h>

typedef __attribute__((ext_vector_type(16))) _Float16 v16h;
typedef __attribute__((ext_vector_type(8)))  float    v8f;

#define BATCH 2
#define NNEUR 512
#define TLEN  1024
#define NM    32     // N_MODES
#define DM    128    // D_MODEL
#define HID   256    // HIDDEN

// ---------------------------------------------------------------------------
// Stage 1a: DFT basis F[t, 2k] = cos(2*pi*k*t/1024), F[t,2k+1] = -sin(...)
// (rfft real/imag parts). Phase reduced mod 1024 exactly in integers.
// ---------------------------------------------------------------------------
__global__ void k_basis(float* __restrict__ F) {
    int idx = blockIdx.x * blockDim.x + threadIdx.x;   // < 1024*64
    int t = idx >> 6;
    int j = idx & 63;
    int k = j >> 1;
    int ph = (k * t) & (TLEN - 1);
    float ang = (float)ph * 6.1359231515425649e-3f;    // 2*pi/1024
    float s, c;
    sincosf(ang, &s, &c);
    F[idx] = (j & 1) ? -s : c;
}

// ---------------------------------------------------------------------------
// Stage 1b: m2[row,k] = xr^2 + xi^2 for first 32 rfft bins. One block per row.
// ---------------------------------------------------------------------------
__global__ __launch_bounds__(256) void k_dft_m2(const float* __restrict__ spikes,
                                                const float* __restrict__ F,
                                                float* __restrict__ m2) {
    __shared__ float sp[TLEN];
    __shared__ float red[256];
    int r = blockIdx.x;
    int tid = threadIdx.x;
    const float* srow = spikes + (size_t)r * TLEN;
    for (int i = tid; i < TLEN; i += 256) sp[i] = srow[i];
    __syncthreads();
    int j = tid & 63;        // output component (2k or 2k+1)
    int c = tid >> 6;        // t-chunk 0..3
    int t0 = c * 256;
    float p = 0.0f;
    for (int t = 0; t < 256; ++t) p += sp[t0 + t] * F[(t0 + t) * 64 + j];
    red[tid] = p;            // tid == c*64 + j
    __syncthreads();
    if (tid < 64) red[tid] = red[tid] + red[tid + 64] + red[tid + 128] + red[tid + 192];
    __syncthreads();
    if (tid < NM) {
        float xr = red[2 * tid], xi = red[2 * tid + 1];
        m2[(size_t)r * NM + tid] = xr * xr + xi * xi;
    }
}

// ---------------------------------------------------------------------------
// Stage 1c: E0[row,d] = mean_k sqrt(m2[row,k]*(Wr^2+Wi^2)[k,d] + 1e-8)
// (|x*w| = |x||w| identity removes the (B,N,32,128) complex tensor entirely)
// ---------------------------------------------------------------------------
__global__ __launch_bounds__(128) void k_E0(const float* __restrict__ m2,
                                            const float* __restrict__ Wr,
                                            const float* __restrict__ Wi,
                                            float* __restrict__ E0) {
    __shared__ float m2r[NM];
    int r = blockIdx.x;
    int d = threadIdx.x;
    if (d < NM) m2r[d] = m2[(size_t)r * NM + d];
    __syncthreads();
    float acc = 0.0f;
    #pragma unroll 4
    for (int k = 0; k < NM; ++k) {
        float wr = Wr[k * DM + d], wi = Wi[k * DM + d];
        acc += sqrtf(m2r[k] * (wr * wr + wi * wi) + 1e-8f);
    }
    E0[(size_t)r * DM + d] = acc * (1.0f / NM);
}

// ---------------------------------------------------------------------------
// Stage 1d: generic per-row MLP layer: out[row,n] = act(sum_k in[row,k]W[k,n]+b[n])
// Small (<0.5 GFLOP total across all uses) -> exact f32 VALU, no precision loss
// where error would compound through the chained MLPs.
// ---------------------------------------------------------------------------
__global__ void k_rowmlp(const float* __restrict__ in, const float* __restrict__ W,
                         const float* __restrict__ bias, float* __restrict__ out,
                         int K, int N, int do_relu) {
    __shared__ float row[HID];
    int r = blockIdx.x;
    int n = threadIdx.x;
    for (int i = n; i < K; i += blockDim.x) row[i] = in[(size_t)r * K + i];
    __syncthreads();
    float acc = bias ? bias[n] : 0.0f;
    #pragma unroll 4
    for (int k = 0; k < K; ++k) acc += row[k] * W[(size_t)k * N + n];
    if (do_relu) acc = acc > 0.0f ? acc : 0.0f;
    out[(size_t)r * N + n] = acc;
}

// ---------------------------------------------------------------------------
// Stage 2: fused pairwise edge kernel (the 34-GFLOP part).
// Workgroup = 8 waves; wave owns one s row and loops 4 tiles of 16 t's.
// Per tile: h1 = relu(a_s + a_t + em_b1) built in registers (f32), packed to
// the ISA-documented f16 A-fragment layout; em_W2 lives in LDS pre-swizzled
// into exact B-fragment order; 8x8 v_wmma_f32_16x16x32_f16 per tile.
// Weight head (rank-1) done in VALU alongside, reduced with shfl_xor.
// ---------------------------------------------------------------------------
__global__ __launch_bounds__(256) void k_edge(
    const float* __restrict__ a_s, const float* __restrict__ a_t,
    const float* __restrict__ w_s, const float* __restrict__ w_t,
    const float* __restrict__ em_b1, const float* __restrict__ em_W2,
    const float* __restrict__ em_b2, const float* __restrict__ em_W3,
    const float* __restrict__ em_b3p,
    const float* __restrict__ wm_b1, const float* __restrict__ wm_W2,
    const float* __restrict__ wm_b2p,
    float* __restrict__ out_logits, float* __restrict__ out_prob,
    float* __restrict__ out_w) {
    // B-fragments: [kt][nt][lane][16 halves], element j of lane l is
    // em_W2[ kt*32 + (l>>4)*8 + (j&7) + (j>>3)*16 , nt*16 + (l&15) ]  (ISA 7.12.2 B layout)
    __shared__ _Float16 ldsB[8 * 8 * 32 * 16];   // 64 KB
    const int tid = threadIdx.x;
    for (int idx = tid; idx < 8 * 8 * 32 * 16; idx += 256) {
        int j    = idx & 15;
        int lane = (idx >> 4) & 31;
        int nt   = (idx >> 9) & 7;
        int kt   = (idx >> 12) & 7;
        int k = kt * 32 + (lane >> 4) * 8 + (j & 7) + (j >> 3) * 16;
        int n = nt * 16 + (lane & 15);
        ldsB[idx] = (_Float16)em_W2[k * DM + n];
    }
    __syncthreads();
    const v16h* Bv = (const v16h*)ldsB;

    int bid  = blockIdx.x;           // 1024 blocks: [b:1][sb:6][tb:3]
    int tb   = bid & 7;              // t super-block of 64
    int sb   = (bid >> 3) & 63;      // s block of 8
    int b    = bid >> 9;
    int wave = tid >> 5;
    int lane = tid & 31;
    int col  = lane & 15;            // pair row (M) / N column / t offset
    int hi   = lane >> 4;
    int kbase = hi * 8;              // A fragment: lanes 16..31 hold K+8 half
    int s = sb * 8 + wave;

    const float em_b3v = em_b3p[0];
    const float wm_b2v = wm_b2p[0];
    const int rs = (b * NNEUR + s) * HID;

    for (int ti = 0; ti < 4; ++ti) {
        int t0 = tb * 64 + ti * 16;
        int t  = t0 + col;                     // this lane's pair row
        const int rt = (b * NNEUR + t) * HID;

        v8f acc[8];
        #pragma unroll
        for (int nt = 0; nt < 8; ++nt) {
            v8f z = {0.f, 0.f, 0.f, 0.f, 0.f, 0.f, 0.f, 0.f};
            acc[nt] = z;
        }
        float wdot = 0.0f;

        for (int kt = 0; kt < 8; ++kt) {
            int klo = kt * 32 + kbase;         // K = klo..klo+7
            int khi = klo + 16;                // K = khi..khi+7
            // h1 = relu(a_s + a_t + em_b1) -> A fragment (16x32 f16 layout)
            float h1v[16];
            #pragma unroll
            for (int j = 0; j < 8; ++j) {
                float v = a_s[rs + klo + j] + a_t[rt + klo + j] + em_b1[klo + j];
                h1v[j] = v > 0.0f ? v : 0.0f;
            }
            #pragma unroll
            for (int j = 0; j < 8; ++j) {
                float v = a_s[rs + khi + j] + a_t[rt + khi + j] + em_b1[khi + j];
                h1v[8 + j] = v > 0.0f ? v : 0.0f;
            }
            v16h af;
            #pragma unroll
            for (int j = 0; j < 16; ++j) af[j] = (_Float16)h1v[j];

            // weight head: lanes l / l+16 cover disjoint K halves of each tile
            #pragma unroll
            for (int j = 0; j < 8; ++j) {
                float v = w_s[rs + klo + j] + w_t[rt + klo + j] + wm_b1[klo + j];
                wdot += (v > 0.0f ? v : 0.0f) * wm_W2[klo + j];
                float v2 = w_s[rs + khi + j] + w_t[rt + khi + j] + wm_b1[khi + j];
                wdot += (v2 > 0.0f ? v2 : 0.0f) * wm_W2[khi + j];
            }

            #pragma unroll
            for (int nt = 0; nt < 8; ++nt) {
                v16h bf = Bv[(kt * 8 + nt) * 32 + lane];
                acc[nt] = __builtin_amdgcn_wmma_f32_16x16x32_f16(
                    false, af, false, bf, (short)0, acc[nt], false, false);
            }
        }

        // h2 = relu(acc + em_b2); logits = h2 @ em_W3 + em_b3
        float lp[8];
        #pragma unroll
        for (int r = 0; r < 8; ++r) lp[r] = 0.0f;
        #pragma unroll
        for (int nt = 0; nt < 8; ++nt) {
            int n = nt * 16 + col;
            float b2 = em_b2[n];
            float w3 = em_W3[n];
            #pragma unroll
            for (int r = 0; r < 8; ++r) {
                float h2 = acc[nt][r] + b2;
                h2 = h2 > 0.0f ? h2 : 0.0f;
                lp[r] += h2 * w3;
            }
        }
        const int obase = (b * NNEUR + s) * NNEUR + t0;
        #pragma unroll
        for (int r = 0; r < 8; ++r) {
            float v = lp[r];                       // pair p = r + 8*hi
            v += __shfl_xor(v, 1, 32);
            v += __shfl_xor(v, 2, 32);
            v += __shfl_xor(v, 4, 32);
            v += __shfl_xor(v, 8, 32);
            if (col == r) {
                float logit = v + em_b3v;
                int oidx = obase + r + 8 * hi;
                out_logits[oidx] = logit;
                out_prob[oidx]   = 1.0f / (1.0f + expf(-logit));
            }
        }
        float wtot = wdot + __shfl_xor(wdot, 16, 32);
        if (hi == 0) {
            float x = wtot + wm_b2v;
            out_w[obase + col] = (x > 20.0f) ? x : log1pf(expf(x));
        }
    }
}

// ---------------------------------------------------------------------------
extern "C" void kernel_launch(void* const* d_in, const int* in_sizes, int n_in,
                              void* d_out, int out_size, void* d_ws, size_t ws_size,
                              hipStream_t stream) {
    (void)in_sizes; (void)n_in; (void)out_size; (void)ws_size;
    const float* spikes = (const float*)d_in[0];
    const float* Wr     = (const float*)d_in[1];
    const float* Wi     = (const float*)d_in[2];
    const float* pm_W1  = (const float*)d_in[3];
    const float* pm_b1  = (const float*)d_in[4];
    const float* pm_W2  = (const float*)d_in[5];
    const float* pm_b2  = (const float*)d_in[6];
    const float* sp_W   = (const float*)d_in[7];
    const float* sp_b   = (const float*)d_in[8];
    const float* tp_W   = (const float*)d_in[9];
    const float* tp_b   = (const float*)d_in[10];
    const float* em_W1  = (const float*)d_in[11];
    const float* em_b1  = (const float*)d_in[12];
    const float* em_W2  = (const float*)d_in[13];
    const float* em_b2  = (const float*)d_in[14];
    const float* em_W3  = (const float*)d_in[15];
    const float* em_b3  = (const float*)d_in[16];
    const float* wm_W1  = (const float*)d_in[17];
    const float* wm_b1  = (const float*)d_in[18];
    const float* wm_W2  = (const float*)d_in[19];
    const float* wm_b2  = (const float*)d_in[20];

    const int R = BATCH * NNEUR;                 // 1024 rows
    float* ws  = (float*)d_ws;
    float* F   = ws;                             // 1024*64
    float* m2  = F   + TLEN * 64;                // 1024*32
    float* E0  = m2  + R * NM;                   // 1024*128
    float* T1  = E0  + R * DM;                   // 1024*128
    float* E   = T1  + R * DM;                   // 1024*128
    float* hs  = E   + R * DM;                   // 1024*256
    float* ht  = hs  + R * HID;
    float* as_ = ht  + R * HID;                  // h_s @ em_W1[:256]
    float* at_ = as_ + R * HID;                  // h_t @ em_W1[256:]
    float* wss = at_ + R * HID;                  // h_s @ wm_W1[:256]
    float* wst = wss + R * HID;                  // h_t @ wm_W1[256:]

    k_basis<<<(TLEN * 64) / 256, 256, 0, stream>>>(F);
    k_dft_m2<<<R, 256, 0, stream>>>(spikes, F, m2);
    k_E0<<<R, 128, 0, stream>>>(m2, Wr, Wi, E0);
    k_rowmlp<<<R, 128, 0, stream>>>(E0, pm_W1, pm_b1, T1, DM, DM, 1);
    k_rowmlp<<<R, 128, 0, stream>>>(T1, pm_W2, pm_b2, E, DM, DM, 0);
    k_rowmlp<<<R, 256, 0, stream>>>(E, sp_W, sp_b, hs, DM, HID, 0);
    k_rowmlp<<<R, 256, 0, stream>>>(E, tp_W, tp_b, ht, DM, HID, 0);
    k_rowmlp<<<R, 256, 0, stream>>>(hs, em_W1,             nullptr, as_, HID, HID, 0);
    k_rowmlp<<<R, 256, 0, stream>>>(ht, em_W1 + HID * HID, nullptr, at_, HID, HID, 0);
    k_rowmlp<<<R, 256, 0, stream>>>(hs, wm_W1,             nullptr, wss, HID, HID, 0);
    k_rowmlp<<<R, 256, 0, stream>>>(ht, wm_W1 + HID * HID, nullptr, wst, HID, HID, 0);

    float* out = (float*)d_out;
    const int P = BATCH * NNEUR * NNEUR;         // 524288 per output
    // grid: 2 (b) * 64 (s blocks of 8) * 8 (t blocks of 64) = 1024 workgroups
    k_edge<<<1024, 256, 0, stream>>>(as_, at_, wss, wst,
                                     em_b1, em_W2, em_b2, em_W3, em_b3,
                                     wm_b1, wm_W2, wm_b2,
                                     out, out + P, out + 2 * P);
}
